// CombineEmbedder_76476187673118
// MI455X (gfx1250) — matrix operand order repaired and checked by reading, compiled, not run on hardware
//
#include <hip/hip_runtime.h>
#include <hip/hip_bf16.h>

#define N_TOTAL 262144
#define FEAT    64
#define DMODEL  128

typedef __attribute__((ext_vector_type(16))) _Float16 v16h;
typedef __attribute__((ext_vector_type(8)))  _Float16 v8h;
typedef __attribute__((ext_vector_type(4)))  _Float16 v4h;
typedef __attribute__((ext_vector_type(2)))  _Float16 v2h;
typedef __attribute__((ext_vector_type(8)))  float    v8f;
typedef __attribute__((ext_vector_type(2)))  float    v2f;
typedef __attribute__((ext_vector_type(4)))  int      v4i;

#if __has_builtin(__builtin_amdgcn_global_load_async_to_lds_b128)
#define HAVE_ASYNC_LDS 1
#else
#define HAVE_ASYNC_LDS 0
#endif

__device__ __forceinline__ float leaky1(float v) { return v >= 0.0f ? v : 0.01f * v; }

// Copy 16B global -> LDS. Async (GLOBAL_LOAD_ASYNC_TO_LDS_B128, ASYNCcnt) if
// the builtin exists, else a plain load+ds_store fallback.
__device__ __forceinline__ void cp16_g2l(_Float16* ldst, const _Float16* gsrc) {
#if HAVE_ASYNC_LDS
    __builtin_amdgcn_global_load_async_to_lds_b128(
        (__attribute__((address_space(1))) v4i*)gsrc,
        (__attribute__((address_space(3))) v4i*)ldst, 0, 0);
#else
    *(uint4*)ldst = *(const uint4*)gsrc;
#endif
}

__device__ __forceinline__ void async_copy_fence() {
#if HAVE_ASYNC_LDS
    asm volatile("s_wait_asynccnt 0x0" ::: "memory");
#endif
}

// Pack an A-matrix fragment (16x32 f16, ISA 7.12.2 layout) from an LDS row.
// lane 0-15 : row = lane,    K = kt*32 + {0..7, 16..23}
// lane 16-31: row = lane-16, K = kt*32 + {8..15, 24..31}
__device__ __forceinline__ v16h pack_a(const _Float16* row_ptr, int kt, int hw) {
    const v8h lo = *(const v8h*)(row_ptr + kt * 32 + hw * 8);
    const v8h hi = *(const v8h*)(row_ptr + kt * 32 + hw * 8 + 16);
    return __builtin_shufflevector(lo, hi, 0, 1, 2, 3, 4, 5, 6, 7,
                                   8, 9, 10, 11, 12, 13, 14, 15);
}

// B fragments pre-packed so each lane reads 16 contiguous halfs (32B) from LDS.
__device__ __forceinline__ v16h load_b(const _Float16* bp, int frag, int lane) {
    return *(const v16h*)(bp + ((size_t)(frag * 32 + lane)) * 16);
}

// ---------------------------------------------------------------------------
// Weight prep: f32 row-major -> f16 WMMA B-fragment layout.
// B (32x16 KxN) layout: lane l holds N = l&15, K = kt*32 + (l>>4)*8 + (i&7) + (i>>3)*16.
// ---------------------------------------------------------------------------
__device__ __forceinline__ void pack_one(const float* W, _Float16* dst, int t) {
    int f = t >> 9;           // fragment id = kt*8 + nt
    int r = t & 511;
    int lane = r >> 4;
    int i = r & 15;
    int kt = f >> 3, nt = f & 7;
    int k = kt * 32 + (lane >> 4) * 8 + (i & 7) + (i >> 3) * 16;
    int n = nt * 16 + (lane & 15);
    dst[t] = (_Float16)W[k * DMODEL + n];
}

__global__ __launch_bounds__(256) void k_prep(const float* We, const float* Wl,
                                              const float* Wn, _Float16* bpe,
                                              _Float16* bpl, _Float16* bpn) {
    int t = blockIdx.x * 256 + threadIdx.x;
    if (t < 8192) {                        // W_embd: 2 kt * 8 nt
        pack_one(We, bpe, t);
    } else if (t < 8192 + 16384) {         // Wl: 4 kt * 8 nt
        pack_one(Wl, bpl, t - 8192);
    } else if (t < 8192 + 16384 + 16384) { // W_node
        pack_one(Wn, bpn, t - 24576);
    }
}

// ---------------------------------------------------------------------------
// Kernel 1: normalize -> embed GEMM -> leaky -> residual MLP -> layernorm*0.5
// One wave per 16-row tile, 8 waves / block (128 rows / block).
// Weights staged in LDS once per block via async copy.
// ---------------------------------------------------------------------------
__global__ __launch_bounds__(256) void k_embed(const float* __restrict__ raw,
                                               const float* __restrict__ smean,
                                               const float* __restrict__ sstd,
                                               const _Float16* __restrict__ bpe,
                                               const float* __restrict__ be,
                                               const _Float16* __restrict__ bpl,
                                               const float* __restrict__ blv,
                                               float* __restrict__ xout) {
    __shared__ _Float16 smF[8 * 16 * FEAT];    // 16 KB feature staging
    __shared__ _Float16 smX[8 * 16 * DMODEL];  // 32 KB x1 staging
    __shared__ _Float16 swE[8192];             // 16 KB W_embd fragments
    __shared__ _Float16 swL[16384];            // 32 KB Wl fragments
    const int tid = threadIdx.x;
    const int lane = tid & 31;
    const int wv = tid >> 5;
    const int rowbase = blockIdx.x * 128 + wv * 16;
    _Float16* sf = smF + wv * 16 * FEAT;
    _Float16* sx = smX + wv * 16 * DMODEL;
    const int hw = lane >> 4;
    const int arow = lane & 15;

    // Stage packed weights into LDS (async): 1024 + 2048 16B chunks.
    for (int i = tid; i < 1024; i += 256) cp16_g2l(swE + i * 8, bpe + i * 8);
    for (int i = tid; i < 2048; i += 256) cp16_g2l(swL + i * 8, bpl + i * 8);

    // Stage normalized f16 features: lane covers cols 2*lane, 2*lane+1.
    // raw_feats is read-once (64 MB): non-temporal so it doesn't evict x in L2.
    const float m0 = smean[2 * lane], m1 = smean[2 * lane + 1];
    const float s0 = sstd[2 * lane] + 1e-3f, s1 = sstd[2 * lane + 1] + 1e-3f;
    for (int m = 0; m < 16; ++m) {
        const v2f rv = __builtin_nontemporal_load(
            ((const v2f*)(raw + (size_t)(rowbase + m) * FEAT)) + lane);
        v2h h;
        h[0] = (_Float16)((rv[0] - m0) / s0);
        h[1] = (_Float16)((rv[1] - m1) / s1);
        ((v2h*)(sf + m * FEAT))[lane] = h;
    }
    async_copy_fence();
    __syncthreads();

    // GEMM 1: (16x64) @ (64x128), K tiles of 32.
    v8f acc1[8] = {};
    for (int kt = 0; kt < 2; ++kt) {
        const v16h a = pack_a(sf + arow * FEAT, kt, hw);
        for (int nt = 0; nt < 8; ++nt) {
            const v16h b = load_b(swE, kt * 8 + nt, lane);
            acc1[nt] = __builtin_amdgcn_wmma_f32_16x16x32_f16(
                false, a, false, b, (short)0, acc1[nt], false, false);
        }
    }
    // bias + leaky -> x1 (kept in registers f32, mirrored f16 to LDS for GEMM2)
    for (int nt = 0; nt < 8; ++nt) {
        const float bb = be[nt * 16 + (lane & 15)];
        const int col = nt * 16 + (lane & 15);
        for (int r = 0; r < 8; ++r) {
            float v = leaky1(acc1[nt][r] + bb);
            acc1[nt][r] = v;
            sx[(r + 8 * hw) * DMODEL + col] = (_Float16)v;
        }
    }
    __syncthreads();

    // GEMM 2: (16x128) @ (128x128)
    v8f acc2[8] = {};
    for (int kt = 0; kt < 4; ++kt) {
        const v16h a = pack_a(sx + arow * DMODEL, kt, hw);
        for (int nt = 0; nt < 8; ++nt) {
            const v16h b = load_b(swL, kt * 8 + nt, lane);
            acc2[nt] = __builtin_amdgcn_wmma_f32_16x16x32_f16(
                false, a, false, b, (short)0, acc2[nt], false, false);
        }
    }
    // x = leaky(leaky(h + bl)) + x1
    for (int nt = 0; nt < 8; ++nt) {
        const float bb = blv[nt * 16 + (lane & 15)];
        for (int r = 0; r < 8; ++r) {
            float v = leaky1(leaky1(acc2[nt][r] + bb));
            acc2[nt][r] = v + acc1[nt][r];
        }
    }
    // layernorm per row (cross-lane reduce inside each 16-lane half) * 0.5
    for (int r = 0; r < 8; ++r) {
        float sum = 0.0f, sq = 0.0f;
        for (int nt = 0; nt < 8; ++nt) {
            const float v = acc2[nt][r];
            sum += v; sq += v * v;
        }
        for (int m = 1; m < 16; m <<= 1) {
            sum += __shfl_xor(sum, m, 32);
            sq  += __shfl_xor(sq,  m, 32);
        }
        const float mu = sum * (1.0f / 128.0f);
        const float var = sq * (1.0f / 128.0f) - mu * mu;
        const float rs = rsqrtf(var + 1e-5f);
        for (int nt = 0; nt < 8; ++nt)
            acc2[nt][r] = (acc2[nt][r] - mu) * rs * 0.5f;
    }
    // store f32
    for (int nt = 0; nt < 8; ++nt) {
        const int col = nt * 16 + (lane & 15);
        for (int r = 0; r < 8; ++r)
            xout[(size_t)(rowbase + r + 8 * hw) * DMODEL + col] = acc2[nt][r];
    }
}

// ---------------------------------------------------------------------------
// Kernel 2 (x2): gather-mean -> GEMM (W_node) -> leaky -> scaled residual add.
// W_node fragments staged in LDS once per block via async copy.
// ---------------------------------------------------------------------------
__global__ __launch_bounds__(256) void k_depth(const float* __restrict__ xsrc,
                                               const int* __restrict__ idmap,
                                               const _Float16* __restrict__ bpn,
                                               const float* __restrict__ bn,
                                               const float* __restrict__ rez,
                                               float* __restrict__ xdst) {
    __shared__ _Float16 smX[8 * 16 * DMODEL];  // 32 KB gathered feats
    __shared__ _Float16 swN[16384];            // 32 KB W_node fragments
    const int tid = threadIdx.x;
    const int lane = tid & 31;
    const int wv = tid >> 5;
    const int rowbase = blockIdx.x * 128 + wv * 16;
    _Float16* sx = smX + wv * 16 * DMODEL;
    const int hw = lane >> 4;
    const int arow = lane & 15;

    for (int i = tid; i < 2048; i += 256) cp16_g2l(swN + i * 8, bpn + i * 8);

    for (int m = 0; m < 16; ++m) {
        const int grow = rowbase + m;
        const int ida = idmap[2 * grow];
        const int idb = idmap[2 * grow + 1];
        if (m + 1 < 16) {  // warm the L2-resident gather rows for next iter
            __builtin_prefetch(xsrc + (size_t)idmap[2 * (grow + 1)] * DMODEL, 0, 0);
            __builtin_prefetch(xsrc + (size_t)idmap[2 * (grow + 1) + 1] * DMODEL, 0, 0);
        }
        const float4 fa = ((const float4*)(xsrc + (size_t)ida * DMODEL))[lane];
        const float4 fb = ((const float4*)(xsrc + (size_t)idb * DMODEL))[lane];
        v4h h;
        h[0] = (_Float16)((fa.x + fb.x) * 0.5f);
        h[1] = (_Float16)((fa.y + fb.y) * 0.5f);
        h[2] = (_Float16)((fa.z + fb.z) * 0.5f);
        h[3] = (_Float16)((fa.w + fb.w) * 0.5f);
        ((v4h*)(sx + m * DMODEL))[lane] = h;
    }
    async_copy_fence();
    __syncthreads();

    v8f acc[8] = {};
    for (int kt = 0; kt < 4; ++kt) {
        const v16h a = pack_a(sx + arow * DMODEL, kt, hw);
        for (int nt = 0; nt < 8; ++nt) {
            const v16h b = load_b(swN, kt * 8 + nt, lane);
            acc[nt] = __builtin_amdgcn_wmma_f32_16x16x32_f16(
                false, a, false, b, (short)0, acc[nt], false, false);
        }
    }
    const float scale = 0.25f * rez[0];  // SCALE_STEPS * rezero
    for (int nt = 0; nt < 8; ++nt) {
        const int col = nt * 16 + (lane & 15);
        const float bb = bn[col];
        for (int r = 0; r < 8; ++r) {
            const float v = leaky1(acc[nt][r] + bb);
            const size_t gi = (size_t)(rowbase + r + 8 * hw) * DMODEL + col;
            xdst[gi] = xsrc[gi] + v * scale;
        }
    }
}

// ---------------------------------------------------------------------------
// Kernel 3: w = x @ W_w, v = x @ W_v. One wave per row.
// ---------------------------------------------------------------------------
__global__ __launch_bounds__(256) void k_wv(const float* __restrict__ xmain,
                                            const float* __restrict__ Ww,
                                            const float* __restrict__ Wv,
                                            float* __restrict__ wout,
                                            float* __restrict__ vout) {
    const int row = (blockIdx.x * 256 + threadIdx.x) >> 5;
    const int lane = threadIdx.x & 31;
    const float4 xv = ((const float4*)(xmain + (size_t)row * DMODEL))[lane];
    const float4 w4 = ((const float4*)Ww)[lane];
    const float4 v4 = ((const float4*)Wv)[lane];
    float dw = xv.x * w4.x + xv.y * w4.y + xv.z * w4.z + xv.w * w4.w;
    float dv = xv.x * v4.x + xv.y * v4.y + xv.z * v4.z + xv.w * v4.w;
    for (int m = 1; m < 32; m <<= 1) {
        dw += __shfl_xor(dw, m, 32);
        dv += __shfl_xor(dv, m, 32);
    }
    if (lane == 0) {
        wout[row] = dw;
        vout[row] = dv;
    }
}

extern "C" void kernel_launch(void* const* d_in, const int* in_sizes, int n_in,
                              void* d_out, int out_size, void* d_ws, size_t ws_size,
                              hipStream_t stream) {
    const float* raw   = (const float*)d_in[0];
    const int*   idmap = (const int*)d_in[2];   // (N,1,2)
    const float* smean = (const float*)d_in[3];
    const float* sstd  = (const float*)d_in[4];
    const float* We    = (const float*)d_in[5];
    const float* be    = (const float*)d_in[6];
    const float* Wl    = (const float*)d_in[7];  // (1,128,128)
    const float* blv   = (const float*)d_in[8];
    const float* Wn    = (const float*)d_in[9];
    const float* bn    = (const float*)d_in[10];
    const float* rez   = (const float*)d_in[11];
    const float* Ww    = (const float*)d_in[12];
    const float* Wv    = (const float*)d_in[13];

    char* ws = (char*)d_ws;
    const size_t XBYTES = (size_t)N_TOTAL * DMODEL * sizeof(float);
    float* xA = (float*)ws;
    float* xB = (float*)(ws + XBYTES);
    _Float16* bpe = (_Float16*)(ws + 2 * XBYTES);
    _Float16* bpl = bpe + 8192;
    _Float16* bpn = bpl + 16384;

    float* xmain = (float*)d_out;                     // N*128
    float* wout  = xmain + (size_t)N_TOTAL * DMODEL;  // N
    float* vout  = wout + N_TOTAL;                    // N

    k_prep <<<160, 256, 0, stream>>>(We, Wl, Wn, bpe, bpl, bpn);
    k_embed<<<N_TOTAL / 128, 256, 0, stream>>>(raw, smean, sstd, bpe, be, bpl, blv, xA);
    k_depth<<<N_TOTAL / 128, 256, 0, stream>>>(xA, idmap, bpn, bn, rez, xB);
    k_depth<<<N_TOTAL / 128, 256, 0, stream>>>(xB, idmap, bpn, bn, rez, xmain);
    k_wv   <<<N_TOTAL / 8, 256, 0, stream>>>(xmain, Ww, Wv, wout, vout);
}
